// AttentionNet_26310969655960
// MI455X (gfx1250) — matrix operand order repaired and checked by reading, not compiled, not run on hardware
//
#include <hip/hip_runtime.h>

// ---------------------------------------------------------------------------
// AttentionNet forward for MI455X (gfx1250, wave32, WMMA bf16)
// ---------------------------------------------------------------------------
// B=64, L=2000, C_IN=4, F=256, K=13, PAD=6, POOL=4, H=8, DH=64, N=500 (pad 512)
// DM=100 (pad 112), NCLS=2
// ---------------------------------------------------------------------------

typedef __attribute__((ext_vector_type(16))) __bf16 v16bf;
typedef __attribute__((ext_vector_type(8)))  float  v8f;

union FragBF { v16bf v; unsigned u[8]; };

__device__ __forceinline__ unsigned short f2bf(float x) {
  unsigned u = __builtin_bit_cast(unsigned, x);
  unsigned r = u + 0x7fffu + ((u >> 16) & 1u);
  return (unsigned short)(r >> 16);
}

// A fragment 16x32 bf16 (ISA 7.12.2): lane m = lane&15,
// lanes 0-15: K 0-7 & 16-23 ; lanes 16-31: K 8-15 & 24-31 (pairs per dword)
__device__ __forceinline__ void load_frag_a(FragBF& f, const unsigned short* base, int ld) {
  int lane = threadIdx.x & 31;
  int r  = lane & 15;
  int hi = (lane >> 4) * 8;
  const unsigned short* p = base + (size_t)r * ld;
#pragma unroll
  for (int j = 0; j < 8; ++j) {
    int k = ((j < 4) ? 0 : 16) + hi + 2 * (j & 3);
    f.u[j] = *(const unsigned*)(p + k);
  }
}

// B fragment 32x16 bf16 loaded from Bt (row-major [Ncols][K]):
// lane n = lane&15, lanes 0-15: K 0-7 & 16-23 ; lanes 16-31: K 8-15 & 24-31
__device__ __forceinline__ void load_frag_b(FragBF& f, const unsigned short* bt, int ld) {
  int lane = threadIdx.x & 31;
  int n  = lane & 15;
  int hi = (lane >> 4) * 8;
  const unsigned short* p = bt + (size_t)n * ld;
#pragma unroll
  for (int j = 0; j < 8; ++j) {
    f.u[j] = *(const unsigned*)(p + ((j < 4) ? 0 : 16) + hi + 2 * (j & 3));
  }
}

__device__ __forceinline__ v8f wmma_bf16(const FragBF& a, const FragBF& b, v8f c) {
  return __builtin_amdgcn_wmma_f32_16x16x32_bf16(false, a.v, false, b.v, (short)0, c,
                                                 false, false);
}

#define NP    512
#define NVAL  500
#define FCH   256
#define BSZ   64
#define LLEN  2000

// ---------------------------------------------------------------------------
// K0: pack weights -> bf16 transposed layouts
//   wqkvt [1536][256] : o<512 Q(h=o>>6,d=o&63), 512..1023 K, 1024.. V
//   biasqkv[1536], wmt [112][512] (Wm transposed, cols>=100 zero)
// ---------------------------------------------------------------------------
__global__ void pack_kernel(const float* __restrict__ Wq, const float* __restrict__ bq,
                            const float* __restrict__ Wk, const float* __restrict__ bk,
                            const float* __restrict__ Wv, const float* __restrict__ bv,
                            const float* __restrict__ Wm,
                            unsigned short* __restrict__ wqkvt,
                            float* __restrict__ biasqkv,
                            unsigned short* __restrict__ wmt) {
  const int NT = 1536 * 256 + 1536 + 112 * 512;
  for (int i = blockIdx.x * blockDim.x + threadIdx.x; i < NT; i += gridDim.x * blockDim.x) {
    if (i < 1536 * 256) {
      int o = i >> 8, ff = i & 255;
      int seg = o >> 9, oo = o & 511;
      int hh = oo >> 6, d = oo & 63;
      const float* W = (seg == 0) ? Wq : (seg == 1) ? Wk : Wv;
      wqkvt[(size_t)o * 256 + ff] = f2bf(W[((size_t)hh * 256 + ff) * 64 + d]);
    } else if (i < 1536 * 256 + 1536) {
      int o = i - 1536 * 256;
      int seg = o >> 9, oo = o & 511;
      const float* bb = (seg == 0) ? bq : (seg == 1) ? bk : bv;
      biasqkv[o] = bb[oo];
    } else {
      int j = i - (1536 * 256 + 1536);
      int o = j >> 9, ii = j & 511;
      wmt[(size_t)o * 512 + ii] = (o < 100) ? f2bf(Wm[(size_t)ii * 100 + o])
                                            : (unsigned short)0;
    }
  }
}

// ---------------------------------------------------------------------------
// K1: conv partial stats per (b,f): sum and sumsq over L
// ---------------------------------------------------------------------------
__global__ void conv_stats_kernel(const float* __restrict__ x, const float* __restrict__ w,
                                  float* __restrict__ psum, float* __restrict__ psq) {
  int bf = blockIdx.x;
  int b = bf >> 8, f = bf & 255;
  __shared__ float wl[52];
  __shared__ float r1[256], r2[256];
  int tid = threadIdx.x;
  if (tid < 52) wl[tid] = w[f * 52 + tid];
  __syncthreads();
  float s = 0.f, sq = 0.f;
  const float* xb = x + (size_t)b * LLEN * 4;
  for (int l = tid; l < LLEN; l += 256) {
    float acc = 0.f;
#pragma unroll
    for (int kk = 0; kk < 13; ++kk) {
      int lp = l - 6 + kk;
      if (lp >= 0 && lp < LLEN) {
        float4 xv = *(const float4*)(xb + (size_t)lp * 4);
        acc += xv.x * wl[kk] + xv.y * wl[13 + kk] + xv.z * wl[26 + kk] + xv.w * wl[39 + kk];
      }
    }
    s += acc; sq += acc * acc;
  }
  r1[tid] = s; r2[tid] = sq;
  __syncthreads();
  for (int st = 128; st > 0; st >>= 1) {
    if (tid < st) { r1[tid] += r1[tid + st]; r2[tid] += r2[tid + st]; }
    __syncthreads();
  }
  if (tid == 0) { psum[f * 64 + b] = r1[0]; psq[f * 64 + b] = r2[0]; }
}

// ---------------------------------------------------------------------------
// K2: finalize BN stats -> scale/shift per channel
// ---------------------------------------------------------------------------
__global__ void bn_finalize_kernel(const float* __restrict__ psum, const float* __restrict__ psq,
                                   const float* __restrict__ gamma, const float* __restrict__ beta,
                                   float* __restrict__ scale, float* __restrict__ shift) {
  int f = threadIdx.x;
  float s = 0.f, sq = 0.f;
  for (int b = 0; b < 64; ++b) { s += psum[f * 64 + b]; sq += psq[f * 64 + b]; }
  const float inv = 1.0f / (64.0f * 2000.0f);
  float mu = s * inv;
  float var = sq * inv - mu * mu;
  float sc = gamma[f] * rsqrtf(var + 1e-5f);
  scale[f] = sc;
  shift[f] = beta[f] - mu * sc;
}

// ---------------------------------------------------------------------------
// K3: conv recompute + BN + ReLU + MaxPool4 -> h bf16 [B][512][256] (pad rows 0)
// ---------------------------------------------------------------------------
__global__ void conv_pool_kernel(const float* __restrict__ x, const float* __restrict__ w,
                                 const float* __restrict__ scale, const float* __restrict__ shift,
                                 unsigned short* __restrict__ h) {
  int bf = blockIdx.x;
  int b = bf >> 8, f = bf & 255;
  __shared__ float wl[52];
  int tid = threadIdx.x;
  if (tid < 52) wl[tid] = w[f * 52 + tid];
  __syncthreads();
  float sc = scale[f], sh = shift[f];
  const float* xb = x + (size_t)b * LLEN * 4;
  for (int n = tid; n < NP; n += 256) {
    unsigned short outv = 0;
    if (n < NVAL) {
      float mx = 0.f;  // relu output >= 0
#pragma unroll
      for (int p = 0; p < 4; ++p) {
        int l = 4 * n + p;
        float acc = 0.f;
#pragma unroll
        for (int kk = 0; kk < 13; ++kk) {
          int lp = l - 6 + kk;
          if (lp >= 0 && lp < LLEN) {
            float4 xv = *(const float4*)(xb + (size_t)lp * 4);
            acc += xv.x * wl[kk] + xv.y * wl[13 + kk] + xv.z * wl[26 + kk] + xv.w * wl[39 + kk];
          }
        }
        float t = fmaxf(sc * acc + sh, 0.f);
        mx = fmaxf(mx, t);
      }
      outv = f2bf(mx);
    }
    h[((size_t)b * NP + n) * FCH + f] = outv;
  }
}

// ---------------------------------------------------------------------------
// K4: QKV GEMM: [32768 x 256] x [256 x 1536] via WMMA bf16.
// One wave = 32 rows x 64 cols: 2 A-frags x 4 B-frags = 8 WMMA / K-step
// (A reuse x4, B reuse x2 -> 1.5 b128 loads per WMMA). Scatter into q/k/vT.
// ---------------------------------------------------------------------------
__global__ void gemm_qkv_kernel(const unsigned short* __restrict__ h,
                                const unsigned short* __restrict__ wt,
                                const float* __restrict__ bias,
                                unsigned short* __restrict__ q,
                                unsigned short* __restrict__ k,
                                unsigned short* __restrict__ vt) {
  int wid = (blockIdx.x * blockDim.x + threadIdx.x) >> 5;
  int lane = threadIdx.x & 31;
  int rt = wid / 24, ct = wid % 24;  // 1024 row tiles (32 rows), 24 col groups (64 cols)
  int row0 = rt * 32, col0 = ct * 64;

  v8f acc[2][4];
#pragma unroll
  for (int r = 0; r < 2; ++r)
#pragma unroll
    for (int i = 0; i < 4; ++i)
      acc[r][i] = (v8f){0.f, 0.f, 0.f, 0.f, 0.f, 0.f, 0.f, 0.f};

  for (int kb = 0; kb < FCH; kb += 32) {
    FragBF a0, a1, bfr;
    load_frag_a(a0, h + (size_t)row0 * FCH + kb, FCH);
    load_frag_a(a1, h + (size_t)(row0 + 16) * FCH + kb, FCH);
#pragma unroll
    for (int i = 0; i < 4; ++i) {
      load_frag_b(bfr, wt + (size_t)(col0 + i * 16) * FCH + kb, FCH);
      acc[0][i] = wmma_bf16(a0, bfr, acc[0][i]);
      acc[1][i] = wmma_bf16(a1, bfr, acc[1][i]);
    }
  }

  int nloc = lane & 15, hi8 = (lane >> 4) * 8;
#pragma unroll
  for (int rsub = 0; rsub < 2; ++rsub) {
    int rbase = row0 + rsub * 16;
#pragma unroll
    for (int i = 0; i < 4; ++i) {
      int col = col0 + i * 16 + nloc;
      float bi = bias[col];
#pragma unroll
      for (int j = 0; j < 8; ++j) {
        int r = rbase + hi8 + j;
        int bb = r >> 9, nn = r & 511;
        unsigned short bv = f2bf(acc[rsub][i][j] + bi);
        if (col < 512) {
          int hh = col >> 6, d = col & 63;
          q[(((size_t)bb * 8 + hh) * NP + nn) * 64 + d] = bv;
        } else if (col < 1024) {
          int oo = col - 512, hh = oo >> 6, d = oo & 63;
          k[(((size_t)bb * 8 + hh) * NP + nn) * 64 + d] = bv;
        } else {
          int oo = col - 1024, hh = oo >> 6, d = oo & 63;
          vt[(((size_t)bb * 8 + hh) * 64 + d) * NP + nn] = bv;
        }
      }
    }
  }
}

// ---------------------------------------------------------------------------
// K5: fused flash attention. Wave = (b,h,16 query rows).
// S^T = K x Q^T (WMMA, C layout => per-lane single-query rows), online softmax,
// P^T -> LDS -> B fragment, O^T += V^T x P (WMMA). Analytic rel-pos bias:
// maxpooled |j-i|/1999 == (4|j-i|+3)/1999.
// ---------------------------------------------------------------------------
__global__ void attn_kernel(const unsigned short* __restrict__ q,
                            const unsigned short* __restrict__ k,
                            const unsigned short* __restrict__ vt,
                            unsigned short* __restrict__ obf) {
  __shared__ unsigned short plds[8][512];
  int widl = threadIdx.x >> 5;
  int lane = threadIdx.x & 31;
  int wid = blockIdx.x * 8 + widl;
  int qt = wid & 31;     // query tile 0..31
  int bh = wid >> 5;     // 0..511
  int b = bh >> 3, hh = bh & 7;

  const unsigned short* qb = q + (size_t)bh * NP * 64;
  const unsigned short* kb = k + (size_t)bh * NP * 64;
  const unsigned short* vb = vt + (size_t)bh * 64 * NP;
  unsigned short* myp = plds[widl];

  FragBF qf0, qf1;
  load_frag_b(qf0, qb + (size_t)qt * 16 * 64 + 0, 64);
  load_frag_b(qf1, qb + (size_t)qt * 16 * 64 + 32, 64);

  v8f o0 = (v8f){0.f,0.f,0.f,0.f,0.f,0.f,0.f,0.f};
  v8f o1 = o0, o2 = o0, o3 = o0;
  float run_max = -1e30f, run_sum = 0.f;

  int nloc = lane & 15;
  int hi8 = (lane >> 4) * 8;
  int n_g = qt * 16 + nloc;  // query position (pooled index)

  for (int m0 = 0; m0 < NP; m0 += 32) {
    v8f s0 = (v8f){0.f,0.f,0.f,0.f,0.f,0.f,0.f,0.f};
    v8f s1 = s0;
    FragBF kf;
    load_frag_a(kf, kb + (size_t)m0 * 64 + 0, 64);        s0 = wmma_bf16(kf, qf0, s0);
    load_frag_a(kf, kb + (size_t)m0 * 64 + 32, 64);       s0 = wmma_bf16(kf, qf1, s0);
    load_frag_a(kf, kb + (size_t)(m0 + 16) * 64 + 0, 64); s1 = wmma_bf16(kf, qf0, s1);
    load_frag_a(kf, kb + (size_t)(m0 + 16) * 64 + 32, 64);s1 = wmma_bf16(kf, qf1, s1);

    float sv[16];
    float tmax = -1e30f;
#pragma unroll
    for (int j = 0; j < 8; ++j) {
      int m = m0 + hi8 + j;
      float t = s0[j] * 0.125f - (4.0f * fabsf((float)(m - n_g)) + 3.0f) * (1.0f / 1999.0f);
      if (m >= NVAL) t = -1e30f;
      sv[j] = t; tmax = fmaxf(tmax, t);
      m = m0 + 16 + hi8 + j;
      t = s1[j] * 0.125f - (4.0f * fabsf((float)(m - n_g)) + 3.0f) * (1.0f / 1999.0f);
      if (m >= NVAL) t = -1e30f;
      sv[8 + j] = t; tmax = fmaxf(tmax, t);
    }
    tmax = fmaxf(tmax, __shfl_xor(tmax, 16, 32));
    float nm = fmaxf(run_max, tmax);
    float alpha = __expf(run_max - nm);
    float lsum = 0.f;
#pragma unroll
    for (int j = 0; j < 16; ++j) {
      float p = __expf(sv[j] - nm);
      sv[j] = p; lsum += p;
    }
    lsum += __shfl_xor(lsum, 16, 32);
    run_sum = run_sum * alpha + lsum;
    run_max = nm;
    o0 *= alpha; o1 *= alpha; o2 *= alpha; o3 *= alpha;

    // write P^T into LDS as row-major P[nq][mk] (16 x 32) bf16
#pragma unroll
    for (int j = 0; j < 8; ++j) {
      myp[nloc * 32 + hi8 + j]      = f2bf(sv[j]);
      myp[nloc * 32 + 16 + hi8 + j] = f2bf(sv[8 + j]);
    }

    FragBF pf;
    load_frag_b(pf, myp, 32);
    FragBF vf;
    load_frag_a(vf, vb + (size_t)0 * 16 * NP + m0, NP); o0 = wmma_bf16(vf, pf, o0);
    load_frag_a(vf, vb + (size_t)1 * 16 * NP + m0, NP); o1 = wmma_bf16(vf, pf, o1);
    load_frag_a(vf, vb + (size_t)2 * 16 * NP + m0, NP); o2 = wmma_bf16(vf, pf, o2);
    load_frag_a(vf, vb + (size_t)3 * 16 * NP + m0, NP); o3 = wmma_bf16(vf, pf, o3);
  }

  float inv = 1.0f / run_sum;
  size_t rowbase = ((size_t)b * NP + (size_t)qt * 16 + nloc) * 512 + (size_t)hh * 64;
#pragma unroll
  for (int j = 0; j < 8; ++j) {
    obf[rowbase + 0 * 16 + hi8 + j] = f2bf(o0[j] * inv);
    obf[rowbase + 1 * 16 + hi8 + j] = f2bf(o1[j] * inv);
    obf[rowbase + 2 * 16 + hi8 + j] = f2bf(o2[j] * inv);
    obf[rowbase + 3 * 16 + hi8 + j] = f2bf(o3[j] * inv);
  }
}

// ---------------------------------------------------------------------------
// K6: Wm GEMM [32768 x 512] x [512 x 112] fused with relu + masked row-sum.
// One wave = 16 rows x all 112 cols: A fragment loaded once per K-step and
// reused by 7 WMMAs (wmt is L2-resident). Deterministic partials:
// pfeat[(b*32 + rowtile)*112 + col]
// ---------------------------------------------------------------------------
__global__ void gemm_wm_kernel(const unsigned short* __restrict__ obf,
                               const unsigned short* __restrict__ wmt,
                               const float* __restrict__ bm,
                               float* __restrict__ pfeat) {
  int rt = (blockIdx.x * blockDim.x + threadIdx.x) >> 5;   // 2048 row tiles
  int lane = threadIdx.x & 31;
  int row0 = rt * 16;

  v8f acc[7];
#pragma unroll
  for (int i = 0; i < 7; ++i) acc[i] = (v8f){0.f,0.f,0.f,0.f,0.f,0.f,0.f,0.f};

  for (int kb = 0; kb < 512; kb += 32) {
    FragBF af, bfr;
    load_frag_a(af, obf + (size_t)row0 * 512 + kb, 512);
#pragma unroll
    for (int i = 0; i < 7; ++i) {
      load_frag_b(bfr, wmt + (size_t)(i * 16) * 512 + kb, 512);
      acc[i] = wmma_bf16(af, bfr, acc[i]);
    }
  }

  int hi8 = (lane >> 4) * 8;
  int b = row0 >> 9;
  int rti = (row0 >> 4) & 31;
#pragma unroll
  for (int i = 0; i < 7; ++i) {
    int col = i * 16 + (lane & 15);
    float bi = (col < 100) ? bm[col] : 0.f;
    float part = 0.f;
#pragma unroll
    for (int j = 0; j < 8; ++j) {
      int r = row0 + hi8 + j;
      int nn = r & 511;
      float v = fmaxf(acc[i][j] + bi, 0.f);
      if (nn < NVAL) part += v;
    }
    part += __shfl_xor(part, 16, 32);
    if (lane < 16) {
      pfeat[((size_t)b * 32 + rti) * 112 + col] = part;
    }
  }
}

// ---------------------------------------------------------------------------
// K7: head: reduce partials -> feat[100], normalize, @Wo + bo -> out[B][2]
// ---------------------------------------------------------------------------
__global__ void head_kernel(const float* __restrict__ pfeat, const float* __restrict__ Wo,
                            const float* __restrict__ bo, float* __restrict__ out) {
  __shared__ float feat[112];
  __shared__ float red[128];
  __shared__ float smean, sstd;
  int b = blockIdx.x, tid = threadIdx.x;
  if (tid < 112) {
    float s = 0.f;
    const float* p = pfeat + (size_t)b * 32 * 112 + tid;
    for (int r = 0; r < 32; ++r) s += p[r * 112];
    feat[tid] = s;
  }
  __syncthreads();
  float v = (tid < 100) ? feat[tid] : 0.f;
  red[tid] = v;
  __syncthreads();
  for (int st = 64; st > 0; st >>= 1) {
    if (tid < st) red[tid] += red[tid + st];
    __syncthreads();
  }
  if (tid == 0) smean = red[0] * 0.01f;
  __syncthreads();
  float mean = smean;
  float d = (tid < 100) ? (v - mean) : 0.f;
  red[tid] = d * d;
  __syncthreads();
  for (int st = 64; st > 0; st >>= 1) {
    if (tid < st) red[tid] += red[tid + st];
    __syncthreads();
  }
  if (tid == 0) sstd = sqrtf(red[0] * 0.01f);
  __syncthreads();
  float inv = 1.0f / (sstd + 1e-6f);
  if (tid < 100) feat[tid] = (v - mean) * inv;
  __syncthreads();
  if (tid < 2) {
    float o = bo[tid];
    for (int c = 0; c < 100; ++c) o += feat[c] * Wo[c * 2 + tid];
    out[b * 2 + tid] = o;
  }
}

// ---------------------------------------------------------------------------
// Workspace layout (bytes, 256-aligned). Total ~146 MB.
// ---------------------------------------------------------------------------
static const size_t OFF_PSUM    = 0;           //  64K f32 16384
static const size_t OFF_PSQ     = 65536;       //  64K
static const size_t OFF_SCALE   = 131072;      //   1K f32 256
static const size_t OFF_SHIFT   = 132096;      //   1K
static const size_t OFF_H       = 133120;      //  16 MB bf16 32768x256
static const size_t OFF_WQKVT   = 16910336;    // 768K bf16 1536x256
static const size_t OFF_BIASQKV = 17696768;    //   6K f32 1536
static const size_t OFF_WMT     = 17702912;    // 112K bf16 112x512
static const size_t OFF_Q       = 17817600;    //  32 MB bf16 [512bh][512][64]
static const size_t OFF_K       = 51372032;    //  32 MB
static const size_t OFF_VT      = 84926464;    //  32 MB bf16 [512bh][64][512]
static const size_t OFF_OBF     = 118480896;   //  32 MB bf16 32768x512
static const size_t OFF_PFEAT   = 152035328;   // 896K f32 64x32x112

extern "C" void kernel_launch(void* const* d_in, const int* in_sizes, int n_in,
                              void* d_out, int out_size, void* d_ws, size_t ws_size,
                              hipStream_t stream) {
  const float* x      = (const float*)d_in[0];
  const float* conv_w = (const float*)d_in[1];
  const float* gamma  = (const float*)d_in[2];
  const float* beta   = (const float*)d_in[3];
  const float* Wq     = (const float*)d_in[4];
  const float* bq     = (const float*)d_in[5];
  const float* Wk     = (const float*)d_in[6];
  const float* bk     = (const float*)d_in[7];
  const float* Wv     = (const float*)d_in[8];
  const float* bv     = (const float*)d_in[9];
  const float* Wm     = (const float*)d_in[10];
  const float* bm     = (const float*)d_in[11];
  const float* Wo     = (const float*)d_in[12];
  const float* bo     = (const float*)d_in[13];
  float* out = (float*)d_out;

  char* ws = (char*)d_ws;
  float* psum             = (float*)(ws + OFF_PSUM);
  float* psq              = (float*)(ws + OFF_PSQ);
  float* scale            = (float*)(ws + OFF_SCALE);
  float* shift            = (float*)(ws + OFF_SHIFT);
  unsigned short* h       = (unsigned short*)(ws + OFF_H);
  unsigned short* wqkvt   = (unsigned short*)(ws + OFF_WQKVT);
  float* biasqkv          = (float*)(ws + OFF_BIASQKV);
  unsigned short* wmt     = (unsigned short*)(ws + OFF_WMT);
  unsigned short* qbuf    = (unsigned short*)(ws + OFF_Q);
  unsigned short* kbuf    = (unsigned short*)(ws + OFF_K);
  unsigned short* vtbuf   = (unsigned short*)(ws + OFF_VT);
  unsigned short* obf     = (unsigned short*)(ws + OFF_OBF);
  float* pfeat            = (float*)(ws + OFF_PFEAT);

  pack_kernel<<<512, 256, 0, stream>>>(Wq, bq, Wk, bk, Wv, bv, Wm, wqkvt, biasqkv, wmt);
  conv_stats_kernel<<<BSZ * FCH, 256, 0, stream>>>(x, conv_w, psum, psq);
  bn_finalize_kernel<<<1, 256, 0, stream>>>(psum, psq, gamma, beta, scale, shift);
  conv_pool_kernel<<<BSZ * FCH, 256, 0, stream>>>(x, conv_w, scale, shift, h);
  // 1024 row tiles (32 rows) * 24 col groups = 24576 waves / 8 waves per block
  gemm_qkv_kernel<<<3072, 256, 0, stream>>>(h, wqkvt, biasqkv, qbuf, kbuf, vtbuf);
  // 512 bh * 32 qtiles = 16384 waves / 8
  attn_kernel<<<2048, 256, 0, stream>>>(qbuf, kbuf, vtbuf, obf);
  // 2048 row tiles (16 rows x 112 cols) = 2048 waves / 8
  gemm_wm_kernel<<<256, 256, 0, stream>>>(obf, wmt, bm, pfeat);
  head_kernel<<<BSZ, 128, 0, stream>>>(pfeat, Wo, bo, out);
}
